// NCAModel3D_34969623724695
// MI455X (gfx1250) — matrix operand order repaired and checked by reading, not compiled, not run on hardware
//
#include <hip/hip_runtime.h>

#define S    64
#define NCH  16
#define HID  128

typedef __attribute__((ext_vector_type(16))) _Float16 v16h;
typedef __attribute__((ext_vector_type(8)))  _Float16 h8;
typedef __attribute__((ext_vector_type(8)))  float    v8f;
typedef int v4i __attribute__((vector_size(16)));

// Async global->LDS path (CDNA5 GLOBAL_LOAD_ASYNC_TO_LDS_B128), guarded so the
// file still compiles (with the synchronous fallback) if the builtin is absent.
#if defined(__gfx1250__) && __has_builtin(__builtin_amdgcn_global_load_async_to_lds_b128)
#define HAVE_ASYNC_LDS 1
typedef __attribute__((address_space(1))) v4i gv4i;   // global int4
typedef __attribute__((address_space(3))) v4i lv4i;   // LDS int4
#else
#define HAVE_ASYNC_LDS 0
#endif

// Sobel kernel weights (normalized by sum|kx| = 32), cross-correlation taps.
// g=0: kx[a,b,c] = w[a]*w[b]*d[c]; g=1: ky = w[a]*d[b]*w[c]; g=2: kz = d[a]*w[b]*w[c]
__host__ __device__ constexpr float kwgt(int g, int a, int b, int c) {
  float wa = (a == 1) ? 2.f : 1.f;
  float wb = (b == 1) ? 2.f : 1.f;
  float wc = (c == 1) ? 2.f : 1.f;
  float da = (float)(a - 1), db = (float)(b - 1), dc = (float)(c - 1);
  float v = (g == 0) ? wa * wb * dc
          : (g == 1) ? wa * db * wc
                     : da * wb * wc;
  return v * (1.0f / 32.0f);
}

// 8-channel depthwise 3^3 conv from LDS tile (strides: d->6*18*16, h->18*16, w->16).
// Fully unrolled; taps with weight 0 are eliminated at compile time (18 live taps).
template <int G>
__device__ inline void conv8(const float* __restrict__ t, float acc[8]) {
#pragma unroll
  for (int a = 0; a < 3; ++a)
#pragma unroll
    for (int b = 0; b < 3; ++b)
#pragma unroll
      for (int c = 0; c < 3; ++c) {
        const float w = kwgt(G, a, b, c);
        if (w != 0.f) {
          const float* p = t + ((a * 6 + b) * 18 + c) * NCH;
#pragma unroll
          for (int ch = 0; ch < 8; ++ch) acc[ch] += w * p[ch];
        }
      }
}

// Kernel A: perception convs + WMMA MLP + stochastic update -> x2 (workspace)
__global__ __launch_bounds__(256) void nca_update_kernel(
    const float* __restrict__ x, const float* __restrict__ w0,
    const float* __restrict__ b0, const float* __restrict__ w1,
    const float* __restrict__ stoch, float* __restrict__ x2) {
  extern __shared__ unsigned char smemRaw[];
  float*    tile = (float*)smemRaw;                      // 6*6*18*16 f32 = 41472 B
  _Float16* w0h  = (_Float16*)(tile + 6 * 6 * 18 * NCH); // 128*64 f16 = 16384 B
  _Float16* w1h  = w0h + HID * 64;                       // 16*128 f16 = 4096 B
  float*    b0s  = (float*)(w1h + NCH * HID);            // 128 f32
  _Float16* hbuf = (_Float16*)(b0s + HID);               // 8 waves * 16*32 f16

  const int tid = threadIdx.x;
  int blk = blockIdx.x;
  const int W0 = (blk & 3)  * 16; blk >>= 2;
  const int H0 = (blk & 15) * 4;  blk >>= 4;
  const int D0 = (blk & 15) * 4;  blk >>= 4;
  const int bb = blk;

  // Prefetch the stochastic-mask rows this block will read in the epilogue.
  if (tid < 16) {
    const size_t vbp =
        (((size_t)bb * S + (D0 + (tid >> 2))) * S + (H0 + (tid & 3))) * S + W0;
    __builtin_prefetch(stoch + vbp, 0, 0);
  }

  // ---- stage x halo tile (f32, zero-padded SAME boundary) ----
  for (int i = tid; i < 6 * 6 * 18; i += 256) {
    const int wz = i % 18;
    const int t2 = i / 18;
    const int hy = t2 % 6;
    const int dz = t2 / 6;
    const int gd = D0 + dz - 1, gh = H0 + hy - 1, gw = W0 + wz - 1;
    float* ldst = tile + (size_t)i * NCH;
    if ((unsigned)gd < S && (unsigned)gh < S && (unsigned)gw < S) {
      const float* src =
          x + ((((size_t)bb * S + gd) * S + gh) * S + gw) * NCH;
#if HAVE_ASYNC_LDS
      // 64B voxel copied as 4 x async b128; no VGPR round-trip (ASYNCcnt path).
      gv4i* g = (gv4i*)src;
      lv4i* l = (lv4i*)ldst;
      __builtin_amdgcn_global_load_async_to_lds_b128(g, l, 0, 0);
      __builtin_amdgcn_global_load_async_to_lds_b128(g, l, 16, 0);
      __builtin_amdgcn_global_load_async_to_lds_b128(g, l, 32, 0);
      __builtin_amdgcn_global_load_async_to_lds_b128(g, l, 48, 0);
#else
      const float4* s4 = (const float4*)src;
      float4* d4 = (float4*)ldst;
      d4[0] = s4[0]; d4[1] = s4[1]; d4[2] = s4[2]; d4[3] = s4[3];
#endif
    } else {
      const float4 z{0, 0, 0, 0};
      float4* d4 = (float4*)ldst;
      d4[0] = z; d4[1] = z; d4[2] = z; d4[3] = z;
    }
  }
  // ---- weights -> LDS (f16) ----
  for (int i = tid; i < HID * 64; i += 256) w0h[i] = (_Float16)w0[i];
  for (int i = tid; i < NCH * HID; i += 256) w1h[i] = (_Float16)w1[i];
  if (tid < HID) b0s[tid] = b0[tid];
#if HAVE_ASYNC_LDS
#if __has_builtin(__builtin_amdgcn_s_wait_asynccnt)
  __builtin_amdgcn_s_wait_asynccnt(0);
#else
  asm volatile("s_wait_asynccnt 0x0" ::: "memory");
#endif
#endif
  __syncthreads();

  const int lane = tid & 31;
  const int wid  = tid >> 5;
  const int vM   = lane & 15;  // voxel row (M) / matrix column (N) within 16
  const int hiK  = lane >> 4;  // lane half selects K sub-range
  const int cb   = hiK * 8;    // channel base this lane supplies

  for (int rr = 0; rr < 2; ++rr) {
    const int row = wid * 2 + rr;       // 16 rows of 16 voxels per block
    const int d = row >> 2, h = row & 3;

    // Perception for this lane's voxel (M = vM along W): 8 channels each.
    const float* tcorner = tile + ((d * 6 + h) * 18 + vM) * NCH + cb;
    float cx[8], cy[8], cz[8];
#pragma unroll
    for (int e = 0; e < 8; ++e) { cx[e] = 0.f; cy[e] = 0.f; cz[e] = 0.f; }
    conv8<0>(tcorner, cx);
    conv8<1>(tcorner, cy);
    conv8<2>(tcorner, cz);
    const float* xcen = tile + (((d + 1) * 6 + (h + 1)) * 18 + (vM + 1)) * NCH + cb;

    // A-fragments (16-bit A layout): chunk0 = [x | convX], chunk1 = [convY | convZ]
    v16h a0, a1;
#pragma unroll
    for (int e = 0; e < 8; ++e) {
      a0[e]     = (_Float16)xcen[e];
      a0[e + 8] = (_Float16)cx[e];
      a1[e]     = (_Float16)cy[e];
      a1[e + 8] = (_Float16)cz[e];
    }

    v8f dxC = {0.f, 0.f, 0.f, 0.f, 0.f, 0.f, 0.f, 0.f};
    _Float16* hw = hbuf + wid * (16 * 32);  // per-wave h transpose bounce

#pragma unroll
    for (int q = 0; q < 4; ++q) {           // fc1 K-chunks (32 hidden each)
#pragma unroll
      for (int jj = 0; jj < 2; ++jj) {      // two fc0 N-tiles per chunk
        const int j = 2 * q + jj;
        // B-fragment of w0: lane column n = j*16+vM, table-order K (8h+s, 16+8h+s)
        const _Float16* wp = w0h + (size_t)(j * 16 + vM) * 64 + hiK * 8;
        v16h bf0, bf1;
#pragma unroll
        for (int e = 0; e < 8; ++e) {
          bf0[e] = wp[e]; bf0[e + 8] = wp[16 + e];
          bf1[e] = wp[32 + e]; bf1[e + 8] = wp[48 + e];
        }
        v8f c = {0.f, 0.f, 0.f, 0.f, 0.f, 0.f, 0.f, 0.f};
        c = __builtin_amdgcn_wmma_f32_16x16x32_f16(false, a0, false, bf0,
                                                   (short)0, c, false, false);
        c = __builtin_amdgcn_wmma_f32_16x16x32_f16(false, a1, false, bf1,
                                                   (short)0, c, false, false);
        const float bias = b0s[j * 16 + vM];
        _Float16* hcol = hw + jj * 16 + vM;
#pragma unroll
        for (int r = 0; r < 8; ++r) {       // C layout: M = r + 8*hiK, N = vM
          float hv = c[r] + bias;
          hv = hv > 0.f ? hv : 0.f;         // ReLU
          hcol[(r + hiK * 8) * 32] = (_Float16)hv;
        }
      }
      asm volatile("" ::: "memory");        // keep LDS store->load ordered (in-order DS)
      // Re-read h as fc1 A-fragment (lane row M = vM, K = q*32 + ...)
      const _Float16* ha = hw + vM * 32 + hiK * 8;
      const h8 alo = *(const h8*)(ha);
      const h8 ahi = *(const h8*)(ha + 16);
      v16h aH;
#pragma unroll
      for (int e = 0; e < 8; ++e) { aH[e] = alo[e]; aH[e + 8] = ahi[e]; }
      // B-fragment of w1: lane column c = vM, table-order K (8h+s, 16+8h+s)
      const _Float16* wq = w1h + (size_t)vM * HID + q * 32 + hiK * 8;
      v16h bW;
#pragma unroll
      for (int e = 0; e < 8; ++e) { bW[e] = wq[e]; bW[e + 8] = wq[16 + e]; }
      dxC = __builtin_amdgcn_wmma_f32_16x16x32_f16(false, aH, false, bW,
                                                   (short)0, dxC, false, false);
      asm volatile("" ::: "memory");
    }

    // Epilogue: dxC holds dx[voxel M = r+8*hiK][channel vM]
    const size_t vb = (((size_t)bb * S + (D0 + d)) * S + (H0 + h)) * S + W0;
#pragma unroll
    for (int r = 0; r < 8; ++r) {
      const int M = r + hiK * 8;
      const float st = stoch[vb + M];
      const float dv = (st > 0.5f) ? dxC[r] : 0.f;   // FIRE_RATE = 0.5
      const float xo = tile[(((d + 1) * 6 + (h + 1)) * 18 + (M + 1)) * NCH + vM];
      x2[(vb + M) * NCH + vM] = xo + dv;
    }
  }
}

// Kernel B: life gating — (maxpool3(x.alpha)>0.1) & (maxpool3(x2.alpha)>0.1)
__global__ __launch_bounds__(256) void nca_life_kernel(
    const float* __restrict__ x, const float* __restrict__ x2,
    float* __restrict__ out) {
  const int v = blockIdx.x * 256 + threadIdx.x;  // voxel id, < 4*64^3
  const int w = v & 63, h = (v >> 6) & 63, d = (v >> 12) & 63, b = v >> 18;
  float m1 = -1e30f, m2 = -1e30f;
#pragma unroll
  for (int dz = -1; dz <= 1; ++dz) {
    const int dd = d + dz;
    if ((unsigned)dd >= S) continue;
#pragma unroll
    for (int hy = -1; hy <= 1; ++hy) {
      const int hh = h + hy;
      if ((unsigned)hh >= S) continue;
#pragma unroll
      for (int wz = -1; wz <= 1; ++wz) {
        const int ww = w + wz;
        if ((unsigned)ww >= S) continue;
        const size_t idx = ((((size_t)b * S + dd) * S + hh) * S + ww) * NCH + 3;
        m1 = fmaxf(m1, x[idx]);
        m2 = fmaxf(m2, x2[idx]);
      }
    }
  }
  const float life = (m1 > 0.1f && m2 > 0.1f) ? 1.f : 0.f;
  const float4* src = (const float4*)(x2 + (size_t)v * NCH);
  float4* dst = (float4*)(out + (size_t)v * NCH);
#pragma unroll
  for (int j = 0; j < 4; ++j) {
    float4 t = src[j];
    t.x *= life; t.y *= life; t.z *= life; t.w *= life;
    dst[j] = t;
  }
}

extern "C" void kernel_launch(void* const* d_in, const int* in_sizes, int n_in,
                              void* d_out, int out_size, void* d_ws, size_t ws_size,
                              hipStream_t stream) {
  const float* x     = (const float*)d_in[0];  // [4,64,64,64,16]
  const float* w0    = (const float*)d_in[1];  // [128,64]
  const float* b0    = (const float*)d_in[2];  // [128]
  const float* w1    = (const float*)d_in[3];  // [16,128]
  const float* stoch = (const float*)d_in[4];  // [4,64,64,64,1]
  float* out = (float*)d_out;
  float* x2  = (float*)d_ws;                   // 4*64^3*16 f32 = 67 MB scratch

  constexpr size_t SMEM = (size_t)(6 * 6 * 18 * NCH) * 4   // x halo tile f32
                        + (size_t)HID * 64 * 2             // w0 f16
                        + (size_t)NCH * HID * 2            // w1 f16
                        + (size_t)HID * 4                  // b0 f32
                        + (size_t)8 * 16 * 32 * 2;         // h bounce (8 waves)

  const int nvox = 4 * S * S * S;
  nca_update_kernel<<<4096, 256, SMEM, stream>>>(x, w0, b0, w1, stoch, x2);
  nca_life_kernel<<<nvox / 256, 256, 0, stream>>>(x, x2, out);
}
